// Model_79482664780184
// MI455X (gfx1250) — matrix operand order repaired and checked by reading, not compiled, not run on hardware
//
#include <hip/hip_runtime.h>
#include <math.h>

// ---------------------------------------------------------------------------
// Types for CDNA5 WMMA (wave32): A/B = 16 halves (8 VGPRs), C/D = 8 floats.
// ---------------------------------------------------------------------------
typedef __attribute__((ext_vector_type(16))) _Float16 v16h;
typedef __attribute__((ext_vector_type(8)))  _Float16 v8h;
typedef __attribute__((ext_vector_type(8)))  float    v8f;

static __device__ __forceinline__ float actf(float v, int act) {
    if (act == 1) return fmaxf(v, 0.f);
    if (act == 2) return 1.f / (1.f + __expf(-v));
    if (act == 3) return tanhf(v);
    return v;
}

// LDS B-tile layout: column-major [n][k], stride 40 halves (80B, 16B-aligned)
// so a lane's 16-half B fragment is two contiguous ds_load_b128.
#define BS_STRIDE 40

// ---------------------------------------------------------------------------
// Implicit-GEMM convolution:  D[Cout x N] = W[Cout x K] * im2col(X)[K x N]
// Block = 4 waves; each wave owns a 16(Cout) x 32(N) tile (2 accumulators,
// 2 x v_wmma_f32_16x16x32_f16 per 32-K step).  B tile (32K x 32N) is staged
// in LDS (fp32 global -> f16) with ping-pong double buffering; A comes from
// pre-packed padded fp16 weights (two 16B loads per lane per step).
// A lane pattern (16-bit A 16x32): lanes 0-15 hold K 0..15 of row
// M=lane; lanes 16-31 hold K 16..31 of row M=lane-16.
// B lane pattern: lane n (0-15) holds K 0..15 of col n; lane n+16 K 16..31.
// C/D: VGPR r -> row M = r + 8*(lane>=16); col N = lane&15.
// ---------------------------------------------------------------------------
__global__ void __launch_bounds__(128) k_conv_wmma(
    const float* __restrict__ in, int inCt, int inCo, int Hin, int Win,
    const _Float16* __restrict__ wpk, int KPad, int Kact, int KH, int KW,
    const float* __restrict__ bias,
    float* __restrict__ out, int outCt, int outCo, int Cout, int Hout, int Wout,
    int stride, int padH, int padW, int Bn, int act, float scale)
{
    __shared__ _Float16 Bs[2][32 * BS_STRIDE];
    const int tid   = threadIdx.x;
    const int lane  = tid & 31;
    const int wv    = tid >> 5;
    const int laneM = lane & 15;
    const int hiK   = lane >> 4;
    const int nBase  = blockIdx.x * 32;
    const int coBase = blockIdx.y * 64 + wv * 16;
    const int Ntot  = Bn * Hout * Wout;
    const int KHW   = KH * KW;
    const int HWo   = Hout * Wout;

    auto stage = [&](int kbase, _Float16* dst) {
        for (int e = tid; e < 1024; e += 128) {
            int kk = e >> 5, n = e & 31;
            float v = 0.f;
            int kg  = kbase + kk;
            int pos = nBase + n;
            if (kg < Kact && pos < Ntot) {
                int ci = kg / KHW; int r2 = kg - ci * KHW;
                int kh = r2 / KW;  int kw = r2 - kh * KW;
                int b  = pos / HWo; int r3 = pos - b * HWo;
                int ho = r3 / Wout; int wo = r3 - ho * Wout;
                int hi = ho * stride - padH + kh;
                int wi = wo * stride - padW + kw;
                if (hi >= 0 && hi < Hin && wi >= 0 && wi < Win)
                    v = in[(((size_t)b * inCt + inCo + ci) * Hin + hi) * Win + wi];
            }
            dst[n * BS_STRIDE + kk] = (_Float16)v;
        }
    };

    v8f acc0 = {}, acc1 = {};
    stage(0, Bs[0]);
    __syncthreads();
    int buf = 0;
    const size_t arow = (size_t)(coBase + laneM) * KPad + hiK * 16;

    for (int k0 = 0; k0 < KPad; k0 += 32) {
        if (k0 + 32 < KPad) {
            stage(k0 + 32, Bs[buf ^ 1]);
            __builtin_prefetch(wpk + arow + k0 + 32, 0, 3);
        }
        // A fragment: two contiguous 16B loads from packed weights
        v16h a, bv0, bv1;
        {
            const v8h lo  = *(const v8h*)(wpk + arow + k0);
            const v8h hi8 = *(const v8h*)(wpk + arow + k0 + 8);
            #pragma unroll
            for (int i = 0; i < 8; i++) { a[i] = lo[i]; a[i + 8] = hi8[i]; }
        }
        // B fragments: two ds_load_b128 per 16-wide column group
        {
            const _Float16* p0 = &Bs[buf][laneM * BS_STRIDE + hiK * 16];
            const _Float16* p1 = &Bs[buf][(laneM + 16) * BS_STRIDE + hiK * 16];
            const v8h x0 = *(const v8h*)p0, x1 = *(const v8h*)(p0 + 8);
            const v8h y0 = *(const v8h*)p1, y1 = *(const v8h*)(p1 + 8);
            #pragma unroll
            for (int i = 0; i < 8; i++) {
                bv0[i] = x0[i]; bv0[i + 8] = x1[i];
                bv1[i] = y0[i]; bv1[i + 8] = y1[i];
            }
        }
        acc0 = __builtin_amdgcn_wmma_f32_16x16x32_f16(false, a, false, bv0,
                                                      (short)0, acc0, false, false);
        acc1 = __builtin_amdgcn_wmma_f32_16x16x32_f16(false, a, false, bv1,
                                                      (short)0, acc1, false, false);
        __syncthreads();
        buf ^= 1;
    }
    // ---- epilogue: bias, scale, activation, scatter (two column groups) ----
    #pragma unroll
    for (int g = 0; g < 2; g++) {
        int n = nBase + laneM + g * 16;
        if (n >= Ntot) continue;
        int b  = n / HWo; int r3 = n - b * HWo;
        int ho = r3 / Wout; int wo = r3 - ho * Wout;
        #pragma unroll
        for (int r = 0; r < 8; r++) {
            int co = coBase + r + 8 * hiK;
            if (co < Cout) {
                float v = ((g == 0 ? acc0[r] : acc1[r]) + bias[co]) * scale;
                v = actf(v, act);
                out[(((size_t)b * outCt + outCo + co) * Hout + ho) * Wout + wo] = v;
            }
        }
    }
}

// ---------------------------------------------------------------------------
// Correlation GEMM:  corr[b][n][m] = (1/16) * sum_c f1T[b][n][c] * f2[b][c][m]
// Same 16x32-per-wave double-buffered structure; HW=3072, D=256 are exact
// multiples of the tiles so no guards are needed.
// ---------------------------------------------------------------------------
__global__ void __launch_bounds__(128) k_corr_wmma(
    const _Float16* __restrict__ aT,  // [B][HW][D]
    const _Float16* __restrict__ bM,  // [B][D][HW]
    float* __restrict__ corr, int HW, int D, float scale)
{
    __shared__ _Float16 Bs[2][32 * BS_STRIDE];
    const int tid = threadIdx.x, lane = tid & 31, wv = tid >> 5;
    const int laneM = lane & 15, hiK = lane >> 4;
    const int mBase = blockIdx.x * 32;
    const int nBase = blockIdx.y * 64 + wv * 16;
    const int b = blockIdx.z;

    auto stage = [&](int kbase, _Float16* dst) {
        for (int e = tid; e < 1024; e += 128) {
            int kk = e >> 5, m = e & 31;
            dst[m * BS_STRIDE + kk] = bM[((size_t)b * D + kbase + kk) * HW + mBase + m];
        }
    };

    v8f acc0 = {}, acc1 = {};
    stage(0, Bs[0]);
    __syncthreads();
    int buf = 0;
    const size_t arow = ((size_t)b * HW + nBase + laneM) * D + hiK * 16;

    for (int k0 = 0; k0 < D; k0 += 32) {
        if (k0 + 32 < D) {
            stage(k0 + 32, Bs[buf ^ 1]);
            __builtin_prefetch(aT + arow + k0 + 32, 0, 3);
        }
        v16h a, bv0, bv1;
        {
            const v8h lo  = *(const v8h*)(aT + arow + k0);
            const v8h hi8 = *(const v8h*)(aT + arow + k0 + 8);
            #pragma unroll
            for (int i = 0; i < 8; i++) { a[i] = lo[i]; a[i + 8] = hi8[i]; }
        }
        {
            const _Float16* p0 = &Bs[buf][laneM * BS_STRIDE + hiK * 16];
            const _Float16* p1 = &Bs[buf][(laneM + 16) * BS_STRIDE + hiK * 16];
            const v8h x0 = *(const v8h*)p0, x1 = *(const v8h*)(p0 + 8);
            const v8h y0 = *(const v8h*)p1, y1 = *(const v8h*)(p1 + 8);
            #pragma unroll
            for (int i = 0; i < 8; i++) {
                bv0[i] = x0[i]; bv0[i + 8] = x1[i];
                bv1[i] = y0[i]; bv1[i + 8] = y1[i];
            }
        }
        acc0 = __builtin_amdgcn_wmma_f32_16x16x32_f16(false, a, false, bv0,
                                                      (short)0, acc0, false, false);
        acc1 = __builtin_amdgcn_wmma_f32_16x16x32_f16(false, a, false, bv1,
                                                      (short)0, acc1, false, false);
        __syncthreads();
        buf ^= 1;
    }
    #pragma unroll
    for (int g = 0; g < 2; g++) {
        int m = mBase + laneM + g * 16;
        #pragma unroll
        for (int r = 0; r < 8; r++) {
            int nrow = nBase + r + 8 * hiK;
            corr[((size_t)b * HW + nrow) * HW + m] = (g == 0 ? acc0[r] : acc1[r]) * scale;
        }
    }
}

// ---------------------------------------------------------------------------
// Small support kernels
// ---------------------------------------------------------------------------
__global__ void k_pack_w(const float* __restrict__ w, _Float16* __restrict__ o,
                         int Co, int K, int KPad, size_t tot)
{
    size_t i = (size_t)blockIdx.x * blockDim.x + threadIdx.x;
    if (i >= tot) return;
    int co = (int)(i / KPad), k = (int)(i % KPad);
    o[i] = (_Float16)((co < Co && k < K) ? w[(size_t)co * K + k] : 0.f);
}

__global__ void k_pack_f1T(const float* __restrict__ f, _Float16* __restrict__ o,
                           int HW, int D, int Bn)
{
    size_t i = (size_t)blockIdx.x * blockDim.x + threadIdx.x;
    size_t tot = (size_t)Bn * HW * D;
    if (i >= tot) return;
    int b = (int)(i / ((size_t)HW * D));
    size_t r = i - (size_t)b * HW * D;
    int n = (int)(r / D), c = (int)(r % D);
    o[i] = (_Float16)f[((size_t)b * D + c) * HW + n];
}

__global__ void k_pack_cvt(const float* __restrict__ f, _Float16* __restrict__ o, size_t n)
{
    size_t i = (size_t)blockIdx.x * blockDim.x + threadIdx.x;
    if (i < n) o[i] = (_Float16)f[i];
}

__global__ void __launch_bounds__(256) k_inst_norm(float* __restrict__ x, int HW, int act)
{
    __shared__ float s1[256], s2[256];
    size_t base = (size_t)blockIdx.x * HW;
    float a = 0.f, b = 0.f;
    for (int i = threadIdx.x; i < HW; i += 256) { float v = x[base + i]; a += v; b += v * v; }
    s1[threadIdx.x] = a; s2[threadIdx.x] = b; __syncthreads();
    for (int st = 128; st > 0; st >>= 1) {
        if (threadIdx.x < st) { s1[threadIdx.x] += s1[threadIdx.x + st];
                                s2[threadIdx.x] += s2[threadIdx.x + st]; }
        __syncthreads();
    }
    float mean = s1[0] / HW;
    float var  = s2[0] / HW - mean * mean;
    float inv  = rsqrtf(var + 1e-5f);
    for (int i = threadIdx.x; i < HW; i += 256) {
        float v = (x[base + i] - mean) * inv;
        if (act) v = fmaxf(v, 0.f);
        x[base + i] = v;
    }
}

__global__ void k_affine(float* __restrict__ x, const float* __restrict__ g,
                         const float* __restrict__ bb, int C, int HW, int act, size_t tot)
{
    size_t i = (size_t)blockIdx.x * blockDim.x + threadIdx.x;
    if (i >= tot) return;
    int c = (int)((i / HW) % C);
    x[i] = actf(x[i] * g[c] + bb[c], act);
}

__global__ void k_add_act(float* __restrict__ d, const float* __restrict__ a,
                          const float* __restrict__ b, size_t n, int act)
{
    size_t i = (size_t)blockIdx.x * blockDim.x + threadIdx.x;
    if (i < n) d[i] = actf(a[i] + b[i], act);
}

__global__ void k_slice(const float* __restrict__ src, int sCt, int sCo,
                        float* __restrict__ dst, int dCt, int dCo,
                        int C, int HW, int Bn, int act)
{
    size_t i = (size_t)blockIdx.x * blockDim.x + threadIdx.x;
    size_t tot = (size_t)Bn * C * HW;
    if (i >= tot) return;
    int b = (int)(i / ((size_t)C * HW));
    size_t r = i - (size_t)b * C * HW;
    int c = (int)(r / HW), p = (int)(r % HW);
    float v = src[((size_t)b * sCt + sCo + c) * HW + p];
    dst[((size_t)b * dCt + dCo + c) * HW + p] = actf(v, act);
}

__global__ void k_mul_slice(float* __restrict__ dst, int dCt, int dCo,
                            const float* __restrict__ a, const float* __restrict__ h,
                            int C, int HW, int Bn)
{
    size_t i = (size_t)blockIdx.x * blockDim.x + threadIdx.x;
    size_t tot = (size_t)Bn * C * HW;
    if (i >= tot) return;
    int b = (int)(i / ((size_t)C * HW));
    size_t r = i - (size_t)b * C * HW;
    int c = (int)(r / HW), p = (int)(r % HW);
    size_t s = ((size_t)b * C + c) * HW + p;
    dst[((size_t)b * dCt + dCo + c) * HW + p] = a[s] * h[s];
}

__global__ void k_gru(float* __restrict__ h, const float* __restrict__ z,
                      const float* __restrict__ q, size_t n)
{
    size_t i = (size_t)blockIdx.x * blockDim.x + threadIdx.x;
    if (i < n) h[i] = (1.f - z[i]) * h[i] + z[i] * q[i];
}

__global__ void k_zero(float* __restrict__ p, size_t n)
{
    size_t i = (size_t)blockIdx.x * blockDim.x + threadIdx.x;
    if (i < n) p[i] = 0.f;
}

__global__ void k_copy(const float* __restrict__ s, float* __restrict__ d, size_t n)
{
    size_t i = (size_t)blockIdx.x * blockDim.x + threadIdx.x;
    if (i < n) d[i] = s[i];
}

__global__ void k_init_coords(float* __restrict__ c1, float* __restrict__ tg,
                              int W, int HW, int Bn)
{
    size_t i = (size_t)blockIdx.x * blockDim.x + threadIdx.x;
    if (i >= (size_t)Bn * HW) return;
    int b = (int)(i / HW), p = (int)(i % HW);
    float x = (float)(p % W), y = (float)(p / W);
    c1[((size_t)b * 2 + 0) * HW + p] = x;  c1[((size_t)b * 2 + 1) * HW + p] = y;
    tg[((size_t)b * 2 + 0) * HW + p] = x;  tg[((size_t)b * 2 + 1) * HW + p] = y;
}

__global__ void k_motion(float* __restrict__ mo, const float* __restrict__ c1,
                         const float* __restrict__ tg, int W, int HW, int Bn)
{
    size_t i = (size_t)blockIdx.x * blockDim.x + threadIdx.x;
    if (i >= (size_t)Bn * HW) return;
    int b = (int)(i / HW), p = (int)(i % HW);
    float x = (float)(p % W), y = (float)(p / W);
    float cx = c1[((size_t)b * 2 + 0) * HW + p], cy = c1[((size_t)b * 2 + 1) * HW + p];
    float tx = tg[((size_t)b * 2 + 0) * HW + p], ty = tg[((size_t)b * 2 + 1) * HW + p];
    mo[((size_t)b * 4 + 0) * HW + p] = cx - x;
    mo[((size_t)b * 4 + 1) * HW + p] = cy - y;
    mo[((size_t)b * 4 + 2) * HW + p] = tx - cx;
    mo[((size_t)b * 4 + 3) * HW + p] = ty - cy;
}

__global__ void k_pool2(const float* __restrict__ in, float* __restrict__ out,
                        int N, int Hi, int Wi)
{
    int Ho = Hi >> 1, Wo = Wi >> 1;
    size_t i = (size_t)blockIdx.x * blockDim.x + threadIdx.x;
    size_t tot = (size_t)N * Ho * Wo;
    if (i >= tot) return;
    int n = (int)(i / ((size_t)Ho * Wo));
    size_t r = i - (size_t)n * Ho * Wo;
    int ho = (int)(r / Wo), wo = (int)(r % Wo);
    const float* ip = in + (size_t)n * Hi * Wi;
    out[i] = 0.25f * (ip[(2 * ho) * Wi + 2 * wo]     + ip[(2 * ho) * Wi + 2 * wo + 1] +
                      ip[(2 * ho + 1) * Wi + 2 * wo] + ip[(2 * ho + 1) * Wi + 2 * wo + 1]);
}

static __device__ __forceinline__ float bil(const float* img, int Hl, int Wl,
                                            float x, float y)
{
    float x0f = floorf(x), y0f = floorf(y);
    float wx = x - x0f, wy = y - y0f;
    int x0 = (int)x0f, y0 = (int)y0f;
    float v00 = 0.f, v01 = 0.f, v10 = 0.f, v11 = 0.f;
    if (y0 >= 0 && y0 < Hl) {
        if (x0 >= 0 && x0 < Wl)         v00 = img[y0 * Wl + x0];
        if (x0 + 1 >= 0 && x0 + 1 < Wl) v01 = img[y0 * Wl + x0 + 1];
    }
    if (y0 + 1 >= 0 && y0 + 1 < Hl) {
        if (x0 >= 0 && x0 < Wl)         v10 = img[(y0 + 1) * Wl + x0];
        if (x0 + 1 >= 0 && x0 + 1 < Wl) v11 = img[(y0 + 1) * Wl + x0 + 1];
    }
    return v00 * (1.f - wx) * (1.f - wy) + v01 * wx * (1.f - wy)
         + v10 * (1.f - wx) * wy         + v11 * wx * wy;
}

__global__ void k_corr_lookup(const float* __restrict__ p0, const float* __restrict__ p1,
                              const float* __restrict__ p2, const float* __restrict__ p3,
                              const float* __restrict__ coords, float* __restrict__ out,
                              int H, int W, int Bn)
{
    int HW = H * W;
    size_t i = (size_t)blockIdx.x * blockDim.x + threadIdx.x;
    size_t tot = (size_t)Bn * HW * 4 * 81;
    if (i >= tot) return;
    int j = (int)(i % 81); size_t t = i / 81;
    int l = (int)(t % 4);  t /= 4;
    int pos = (int)(t % HW);
    int b   = (int)(t / HW);
    float cx = coords[((size_t)b * 2 + 0) * HW + pos];
    float cy = coords[((size_t)b * 2 + 1) * HW + pos];
    const float* pl = (l == 0) ? p0 : (l == 1) ? p1 : (l == 2) ? p2 : p3;
    int Hl = H >> l, Wl = W >> l;
    float sc = 1.f / (float)(1 << l);
    float dx = (float)(j % 9) - 4.f, dy = (float)(j / 9) - 4.f;
    const float* img = pl + (size_t)(b * HW + pos) * Hl * Wl;
    out[((size_t)b * 324 + l * 81 + j) * HW + pos] =
        bil(img, Hl, Wl, cx * sc + dx, cy * sc + dy);
}

// DLT normal-matrix reduction: M[b] = sum_n w^2 (a1 a1^T + a2 a2^T), 9x9 sym.
__global__ void __launch_bounds__(256) k_dlt(const float* __restrict__ target,
                                             const float* __restrict__ wt,
                                             float* __restrict__ Mg, int W, int HW)
{
    __shared__ float sd[256];
    int b = blockIdx.y, tid = threadIdx.x;
    float m[45];
    #pragma unroll
    for (int t = 0; t < 45; t++) m[t] = 0.f;
    for (int p = blockIdx.x * 256 + tid; p < HW; p += gridDim.x * 256) {
        float x = (float)(p % W), y = (float)(p / W);
        float xp = target[((size_t)b * 2 + 0) * HW + p];
        float yp = target[((size_t)b * 2 + 1) * HW + p];
        float wv = wt[(size_t)b * HW + p];
        float w2 = wv * wv;
        float a1[9] = {x, y, 1.f, 0.f, 0.f, 0.f, -xp * x, -xp * y, -xp};
        float a2[9] = {0.f, 0.f, 0.f, x, y, 1.f, -yp * x, -yp * y, -yp};
        int t = 0;
        #pragma unroll
        for (int ii = 0; ii < 9; ii++)
            #pragma unroll
            for (int jj = ii; jj < 9; jj++) {
                m[t] += w2 * (a1[ii] * a1[jj] + a2[ii] * a2[jj]);
                t++;
            }
    }
    int t = 0;
    #pragma unroll
    for (int ii = 0; ii < 9; ii++)
        #pragma unroll
        for (int jj = ii; jj < 9; jj++) {
            sd[tid] = m[t]; __syncthreads();
            for (int st = 128; st > 0; st >>= 1) {
                if (tid < st) sd[tid] += sd[tid + st];
                __syncthreads();
            }
            if (tid == 0) {
                atomicAdd(&Mg[b * 81 + ii * 9 + jj], sd[0]);
                if (ii != jj) atomicAdd(&Mg[b * 81 + jj * 9 + ii], sd[0]);
            }
            __syncthreads();
            t++;
        }
}

// Per-batch 9x9 symmetric Jacobi eigen-solve; emit eigvec of min eigenvalue,
// normalized so H[2][2] == 1.
__global__ void k_eigen(const float* __restrict__ Mg, float* __restrict__ Hm, int Bn)
{
    int b = threadIdx.x;
    if (b >= Bn) return;
    float A[81], V[81];
    for (int i = 0; i < 81; i++) { A[i] = Mg[b * 81 + i]; V[i] = 0.f; }
    for (int i = 0; i < 9; i++) V[i * 9 + i] = 1.f;
    for (int sweep = 0; sweep < 40; sweep++) {
        for (int p = 0; p < 8; p++)
            for (int q = p + 1; q < 9; q++) {
                float apq = A[p * 9 + q];
                if (fabsf(apq) < 1e-18f) continue;
                float app = A[p * 9 + p], aqq = A[q * 9 + q];
                float th  = 0.5f * (aqq - app) / apq;
                float tt  = ((th >= 0.f) ? 1.f : -1.f) / (fabsf(th) + sqrtf(th * th + 1.f));
                float c = rsqrtf(tt * tt + 1.f);
                float s = tt * c;
                for (int k = 0; k < 9; k++) {
                    float akp = A[k * 9 + p], akq = A[k * 9 + q];
                    A[k * 9 + p] = c * akp - s * akq;
                    A[k * 9 + q] = s * akp + c * akq;
                }
                for (int k = 0; k < 9; k++) {
                    float apk = A[p * 9 + k], aqk = A[q * 9 + k];
                    A[p * 9 + k] = c * apk - s * aqk;
                    A[q * 9 + k] = s * apk + c * aqk;
                }
                for (int k = 0; k < 9; k++) {
                    float vkp = V[k * 9 + p], vkq = V[k * 9 + q];
                    V[k * 9 + p] = c * vkp - s * vkq;
                    V[k * 9 + q] = s * vkp + c * vkq;
                }
            }
    }
    int mi = 0; float mv = A[0];
    for (int i = 1; i < 9; i++) { float d = A[i * 9 + i]; if (d < mv) { mv = d; mi = i; } }
    float h8  = V[8 * 9 + mi];
    float inv = (fabsf(h8) > 1e-30f) ? 1.f / h8 : 0.f;
    for (int i = 0; i < 9; i++) Hm[b * 9 + i] = V[i * 9 + mi] * inv;
}

__global__ void k_project(const float* __restrict__ Hm, const float* __restrict__ target,
                          float* __restrict__ c1, float* __restrict__ flow8,
                          float* __restrict__ resid, int W, int HW, int Bn)
{
    size_t i = (size_t)blockIdx.x * blockDim.x + threadIdx.x;
    if (i >= (size_t)Bn * HW) return;
    int b = (int)(i / HW), p = (int)(i % HW);
    float x = (float)(p % W), y = (float)(p / W);
    const float* h = Hm + b * 9;
    float p0 = h[0] * x + h[1] * y + h[2];
    float p1 = h[3] * x + h[4] * y + h[5];
    float p2 = h[6] * x + h[7] * y + h[8];
    float inv = 1.f / p2;
    float cx = p0 * inv, cy = p1 * inv;
    c1[((size_t)b * 2 + 0) * HW + p] = cx;
    c1[((size_t)b * 2 + 1) * HW + p] = cy;
    flow8[((size_t)b * 2 + 0) * HW + p] = cx - x;
    flow8[((size_t)b * 2 + 1) * HW + p] = cy - y;
    resid[((size_t)b * 2 + 0) * HW + p] = target[((size_t)b * 2 + 0) * HW + p] - cx;
    resid[((size_t)b * 2 + 1) * HW + p] = target[((size_t)b * 2 + 1) * HW + p] - cy;
}

__global__ void k_upsample(const float* __restrict__ flow, const float* __restrict__ mask,
                           float* __restrict__ outF, int H, int W, int Bn)
{
    size_t idx = (size_t)blockIdx.x * blockDim.x + threadIdx.x;
    size_t tot = (size_t)Bn * H * W * 64;
    if (idx >= tot) return;
    int ij = (int)(idx & 63); size_t t = idx >> 6;
    int w = (int)(t % W); t /= W;
    int h = (int)(t % H);
    int b = (int)(t / H);
    int i = ij >> 3, j = ij & 7;
    int HW = H * W;
    float mv[9]; float mx = -1e30f;
    #pragma unroll
    for (int k = 0; k < 9; k++) {
        int ch = (k * 8 + i) * 8 + j;
        float m = mask[((size_t)b * 576 + ch) * HW + h * W + w];
        mv[k] = m; mx = fmaxf(mx, m);
    }
    float sum = 0.f;
    #pragma unroll
    for (int k = 0; k < 9; k++) { mv[k] = __expf(mv[k] - mx); sum += mv[k]; }
    float inv = 1.f / sum;
    int HO = H * 8, WO = W * 8;
    #pragma unroll
    for (int c = 0; c < 2; c++) {
        float acc = 0.f;
        #pragma unroll
        for (int k = 0; k < 9; k++) {
            int di = k / 3 - 1, dj = k % 3 - 1;
            int hh = h + di, ww = w + dj;
            float pv = (hh >= 0 && hh < H && ww >= 0 && ww < W)
                     ? 8.f * flow[((size_t)b * 2 + c) * HW + hh * W + ww] : 0.f;
            acc += mv[k] * pv;
        }
        outF[(((size_t)b * 2 + c) * HO + (h * 8 + i)) * WO + (w * 8 + j)] = acc * inv;
    }
}

// ---------------------------------------------------------------------------
// Host orchestration
// ---------------------------------------------------------------------------
static inline int ru(int x, int m) { return (x + m - 1) / m * m; }

struct CW { const float* w; const float* b; int Ci, Co, KH, KW, K, KPad; _Float16* pk; };
struct NormP { const float* g; const float* b; };
struct BlockP { CW c1, c2; NormP n1, n2, n3; bool ds; CW dsc; };
struct EncP { CW c1; NormP n1; BlockP b[6]; CW c2; bool bn; };

extern "C" void kernel_launch(void* const* d_in, const int* in_sizes, int n_in,
                              void* d_out, int out_size, void* d_ws, size_t ws_size,
                              hipStream_t stream)
{
    (void)in_sizes; (void)n_in; (void)out_size; (void)ws_size;
    const int Bn = 2, H = 48, W = 64, HW = H * W, D = 256, ITERS = 12;

    const float* image1 = (const float*)d_in[0];
    const float* image2 = (const float*)d_in[1];
    int pidx = 2;
    auto F = [&]() { return (const float*)d_in[pidx++]; };

    // -------- deterministic bump allocator over d_ws --------
    char* wsb = (char*)d_ws;
    size_t off = 0;
    auto take = [&](size_t bytes) -> void* {
        off = (off + 255) & ~(size_t)255;
        void* p = wsb + off; off += bytes; return p;
    };
    auto takef = [&](size_t n) { return (float*)take(n * sizeof(float)); };
    auto takeh = [&](size_t n) { return (_Float16*)take(n * sizeof(_Float16)); };

    // -------- fixed buffers --------
    float* pyr0 = takef(18874368);         // also hosts encoder ping-pong bufs
    float* pyr1 = takef(4718592);
    float* pyr2 = takef(1179648);
    float* pyr3 = takef(294912);
    float* EA = pyr0;
    float* EB = pyr0 + 6291456;
    float* EC = pyr0 + 12582912;
    float* f1feat = takef(1572864);
    float* f2feat = takef(1572864);
    float* cnfeat = takef(1572864);
    float* netb = takef(786432);
    float* inpb = takef(786432);
    _Float16* f1T = takeh(1572864);
    _Float16* f2h = takeh(1572864);
    float* corrf = takef(1990656);
    float* motion = takef(24576);
    float* T256 = takef(1572864);
    float* TM128 = takef(786432);
    float* CF = takef(1572864);
    float* MF = takef(786432);
    float* HX = takef(2359296);
    float* QX = takef(2359296);
    float* zb = takef(786432);
    float* rb = takef(786432);
    float* qb = takef(786432);
    float* dfb = takef(12288);
    float* umb = takef(3538944);
    float* wtb = takef(6144);
    float* coords1 = takef(12288);
    float* target = takef(12288);
    float* flow8 = takef(12288);
    float* Mg = takef(162);
    float* Hmb = takef(18);

    // -------- parse params + pack weights to fp16 (A-operand friendly) ------
    auto mkcw = [&](int Ci, int Co, int KH, int KW) -> CW {
        CW c; c.w = F(); c.b = F();
        c.Ci = Ci; c.Co = Co; c.KH = KH; c.KW = KW;
        c.K = Ci * KH * KW; c.KPad = ru(c.K, 32);
        int CoPad = ru(Co, 64);
        size_t tot = (size_t)CoPad * c.KPad;
        c.pk = takeh(tot);
        k_pack_w<<<dim3((unsigned)((tot + 255) / 256)), 256, 0, stream>>>(
            c.w, c.pk, c.Co, c.K, c.KPad, tot);
        return c;
    };
    auto mknorm = [&]() -> NormP { NormP n; n.g = F(); n.b = F(); return n; };

    auto parseEnc = [&](bool bn) -> EncP {
        EncP E; E.bn = bn;
        E.c1 = mkcw(3, 64, 7, 7);
        if (bn) E.n1 = mknorm();
        const int ci_[3] = {64, 64, 96}, co_[3] = {64, 96, 128}, s_[3] = {1, 2, 2};
        int k = 0;
        for (int l = 0; l < 3; l++)
            for (int bi = 0; bi < 2; bi++) {
                int bci = bi == 0 ? ci_[l] : co_[l];
                int bs  = bi == 0 ? s_[l] : 1;
                BlockP& B = E.b[k++];
                B.c1 = mkcw(bci, co_[l], 3, 3);
                B.c2 = mkcw(co_[l], co_[l], 3, 3);
                if (bn) { B.n1 = mknorm(); B.n2 = mknorm(); }
                B.ds = (bci != co_[l]) || (bs != 1);
                if (B.ds) { B.dsc = mkcw(bci, co_[l], 1, 1); if (bn) B.n3 = mknorm(); }
            }
        E.c2 = mkcw(128, 256, 1, 1);
        return E;
    };

    EncP fnet = parseEnc(false);
    EncP cnet = parseEnc(true);
    CW mec1 = mkcw(324, 256, 1, 1), mec2 = mkcw(256, 192, 3, 3);
    CW mef1 = mkcw(4, 128, 7, 7),   mef2 = mkcw(128, 64, 3, 3);
    CW meconv = mkcw(256, 124, 3, 3);
    CW gz1 = mkcw(384, 128, 1, 5), gr1 = mkcw(384, 128, 1, 5), gq1 = mkcw(384, 128, 1, 5);
    CW gz2 = mkcw(384, 128, 5, 1), gr2 = mkcw(384, 128, 5, 1), gq2 = mkcw(384, 128, 5, 1);
    CW fh1 = mkcw(128, 256, 3, 3), fh2 = mkcw(256, 2, 3, 3);
    CW mk1 = mkcw(128, 256, 3, 3), mk2 = mkcw(256, 576, 1, 1);
    CW wh1 = mkcw(128, 256, 3, 3), wh2 = mkcw(256, 1, 3, 3);
    // d_in[pidx] == iters (fixed to 12 by the reference setup; not host-readable
    // inside graph capture)

    // -------- launch helpers --------
    auto conv = [&](const float* in, int inCt, int inCo, int Hin, int Win,
                    const CW& cw, float* out, int outCt, int outCo,
                    int Hout, int Wout, int stride, int pH, int pW,
                    int act, float scale) {
        int Ntot = Bn * Hout * Wout;
        dim3 g((Ntot + 31) / 32, (cw.Co + 63) / 64);
        k_conv_wmma<<<g, 128, 0, stream>>>(in, inCt, inCo, Hin, Win,
                                           cw.pk, cw.KPad, cw.K, cw.KH, cw.KW,
                                           cw.b, out, outCt, outCo, cw.Co,
                                           Hout, Wout, stride, pH, pW, Bn, act, scale);
    };
    auto instn = [&](float* x, int C, int hw, int act) {
        k_inst_norm<<<Bn * C, 256, 0, stream>>>(x, hw, act);
    };
    auto affn = [&](float* x, const NormP& n, int C, int hw, int act) {
        size_t tot = (size_t)Bn * C * hw;
        k_affine<<<(unsigned)((tot + 255) / 256), 256, 0, stream>>>(
            x, n.g, n.b, C, hw, act, tot);
    };
    auto slice = [&](const float* s, int sCt, int sCo, float* d, int dCt, int dCo,
                     int C, int act) {
        size_t tot = (size_t)Bn * C * HW;
        k_slice<<<(unsigned)((tot + 255) / 256), 256, 0, stream>>>(
            s, sCt, sCo, d, dCt, dCo, C, HW, Bn, act);
    };

    // -------- encoders --------
    auto runEnc = [&](const float* img, const EncP& E, float* outFeat) {
        conv(img, 3, 0, 384, 512, E.c1, EA, 64, 0, 192, 256, 2, 3, 3, 0, 1.f);
        if (E.bn) affn(EA, E.n1, 64, 192 * 256, 1); else instn(EA, 64, 192 * 256, 1);
        const int Hi[6] = {192, 192, 192, 96, 96, 48}, Wi[6] = {256, 256, 256, 128, 128, 64};
        const int Ho[6] = {192, 192, 96, 96, 48, 48},  Wo[6] = {256, 256, 128, 128, 64, 64};
        const int ss[6] = {1, 1, 2, 1, 2, 1};
        for (int k = 0; k < 6; k++) {
            const BlockP& B = E.b[k];
            int hwOut = Ho[k] * Wo[k];
            int Cin = B.c1.Ci, Co = B.c1.Co;
            conv(EA, Cin, 0, Hi[k], Wi[k], B.c1, EB, Co, 0, Ho[k], Wo[k], ss[k], 1, 1, 0, 1.f);
            if (E.bn) affn(EB, B.n1, Co, hwOut, 1); else instn(EB, Co, hwOut, 1);
            conv(EB, Co, 0, Ho[k], Wo[k], B.c2, EC, Co, 0, Ho[k], Wo[k], 1, 1, 1, 0, 1.f);
            if (E.bn) affn(EC, B.n2, Co, hwOut, 1); else instn(EC, Co, hwOut, 1);
            size_t n = (size_t)Bn * Co * hwOut;
            if (B.ds) {
                conv(EA, Cin, 0, Hi[k], Wi[k], B.dsc, EB, Co, 0, Ho[k], Wo[k], ss[k], 0, 0, 0, 1.f);
                if (E.bn) affn(EB, B.n3, Co, hwOut, 0); else instn(EB, Co, hwOut, 0);
                k_add_act<<<(unsigned)((n + 255) / 256), 256, 0, stream>>>(EA, EB, EC, n, 1);
            } else {
                k_add_act<<<(unsigned)((n + 255) / 256), 256, 0, stream>>>(EA, EA, EC, n, 1);
            }
        }
        conv(EA, 128, 0, H, W, E.c2, outFeat, 256, 0, H, W, 1, 0, 0, 0, 1.f);
    };

    runEnc(image1, fnet, f1feat);
    runEnc(image2, fnet, f2feat);
    runEnc(image1, cnet, cnfeat);
    slice(cnfeat, 256, 0,   netb, 128, 0, 128, 3); // tanh
    slice(cnfeat, 256, 128, inpb, 128, 0, 128, 1); // relu

    // -------- correlation pyramid (WMMA GEMM) --------
    {
        size_t tot = (size_t)Bn * HW * D;
        k_pack_f1T<<<(unsigned)((tot + 255) / 256), 256, 0, stream>>>(f1feat, f1T, HW, D, Bn);
        k_pack_cvt<<<(unsigned)((tot + 255) / 256), 256, 0, stream>>>(f2feat, f2h, tot);
        k_corr_wmma<<<dim3(HW / 32, HW / 64, Bn), 128, 0, stream>>>(f1T, f2h, pyr0, HW, D, 1.f / 16.f);
        int N = Bn * HW;
        size_t t1 = (size_t)N * 24 * 32, t2 = (size_t)N * 12 * 16, t3 = (size_t)N * 6 * 8;
        k_pool2<<<(unsigned)((t1 + 255) / 256), 256, 0, stream>>>(pyr0, pyr1, N, 48, 64);
        k_pool2<<<(unsigned)((t2 + 255) / 256), 256, 0, stream>>>(pyr1, pyr2, N, 24, 32);
        k_pool2<<<(unsigned)((t3 + 255) / 256), 256, 0, stream>>>(pyr2, pyr3, N, 12, 16);
    }

    k_init_coords<<<(Bn * HW + 255) / 256, 256, 0, stream>>>(coords1, target, W, HW, Bn);

    float* dout = (float*)d_out;
    const size_t flowsPer = (size_t)Bn * 2 * 384 * 512;        // 786432
    const size_t residOff = (size_t)ITERS * flowsPer;          // 9437184
    const size_t residPer = (size_t)Bn * 2 * HW;               // 12288
    const size_t wsOff = residOff + (size_t)ITERS * residPer;  // 9584640
    const size_t wsPer = (size_t)Bn * HW;                      // 6144

    auto gruStep = [&](const CW& cz, const CW& cr, const CW& cq, int pH, int pW) {
        slice(netb, 128, 0, HX, 384, 0, 128, 0);
        conv(HX, 384, 0, H, W, cz, zb, 128, 0, H, W, 1, pH, pW, 2, 1.f);
        conv(HX, 384, 0, H, W, cr, rb, 128, 0, H, W, 1, pH, pW, 2, 1.f);
        size_t n = (size_t)Bn * 128 * HW;
        k_mul_slice<<<(unsigned)((n + 255) / 256), 256, 0, stream>>>(QX, 384, 0, rb, netb, 128, HW, Bn);
        conv(QX, 384, 0, H, W, cq, qb, 128, 0, H, W, 1, pH, pW, 3, 1.f);
        k_gru<<<(unsigned)((n + 255) / 256), 256, 0, stream>>>(netb, zb, qb, n);
    };

    for (int it = 0; it < ITERS; it++) {
        size_t lkTot = (size_t)Bn * HW * 4 * 81;
        k_corr_lookup<<<(unsigned)((lkTot + 255) / 256), 256, 0, stream>>>(
            pyr0, pyr1, pyr2, pyr3, coords1, corrf, H, W, Bn);
        k_motion<<<(Bn * HW + 255) / 256, 256, 0, stream>>>(motion, coords1, target, W, HW, Bn);

        // motion encoder
        conv(corrf, 324, 0, H, W, mec1, T256, 256, 0, H, W, 1, 0, 0, 1, 1.f);
        conv(T256, 256, 0, H, W, mec2, CF, 256, 0,   H, W, 1, 1, 1, 1, 1.f);
        conv(motion, 4, 0, H, W, mef1, TM128, 128, 0, H, W, 1, 3, 3, 1, 1.f);
        conv(TM128, 128, 0, H, W, mef2, CF, 256, 192, H, W, 1, 1, 1, 1, 1.f);
        conv(CF, 256, 0, H, W, meconv, MF, 128, 0, H, W, 1, 1, 1, 1, 1.f);
        slice(motion, 4, 0, MF, 128, 124, 4, 0);

        // x = [inp | mf] into both GRU input buffers
        slice(inpb, 128, 0, HX, 384, 128, 128, 0);
        slice(MF,   128, 0, HX, 384, 256, 128, 0);
        slice(inpb, 128, 0, QX, 384, 128, 128, 0);
        slice(MF,   128, 0, QX, 384, 256, 128, 0);

        gruStep(gz1, gr1, gq1, 0, 2);  // 1x5
        gruStep(gz2, gr2, gq2, 2, 0);  // 5x1

        // heads
        conv(netb, 128, 0, H, W, fh1, T256, 256, 0, H, W, 1, 1, 1, 1, 1.f);
        conv(T256, 256, 0, H, W, fh2, dfb, 2, 0, H, W, 1, 1, 1, 0, 1.f);
        conv(netb, 128, 0, H, W, mk1, T256, 256, 0, H, W, 1, 1, 1, 1, 1.f);
        conv(T256, 256, 0, H, W, mk2, umb, 576, 0, H, W, 1, 0, 0, 0, 0.25f);
        conv(netb, 128, 0, H, W, wh1, T256, 256, 0, H, W, 1, 1, 1, 1, 1.f);
        conv(T256, 256, 0, H, W, wh2, wtb, 1, 0, H, W, 1, 1, 1, 2, 1.f);

        // target = coords1 + df
        k_add_act<<<(unsigned)((residPer + 255) / 256), 256, 0, stream>>>(
            target, coords1, dfb, residPer, 0);

        // weighted DLT -> homography -> reproject
        k_zero<<<1, 256, 0, stream>>>(Mg, 162);
        k_dlt<<<dim3((HW + 255) / 256, Bn), 256, 0, stream>>>(target, wtb, Mg, W, HW);
        k_eigen<<<1, 32, 0, stream>>>(Mg, Hmb, Bn);
        k_project<<<(Bn * HW + 255) / 256, 256, 0, stream>>>(
            Hmb, target, coords1, flow8, dout + residOff + (size_t)it * residPer, W, HW, Bn);

        // outputs
        size_t upTot = (size_t)Bn * H * W * 64;
        k_upsample<<<(unsigned)((upTot + 255) / 256), 256, 0, stream>>>(
            flow8, umb, dout + (size_t)it * flowsPer, H, W, Bn);
        k_copy<<<(unsigned)((wsPer + 255) / 256), 256, 0, stream>>>(
            wtb, dout + wsOff + (size_t)it * wsPer, wsPer);
    }
}
